// GeToEdgeAggregator_63067299774722
// MI455X (gfx1250) — matrix operand order repaired and checked by reading, not compiled, not run on hardware
//
#include <hip/hip_runtime.h>
#include <hip/hip_bf16.h>

typedef __bf16 bf16_t;
typedef __attribute__((ext_vector_type(16))) __bf16 v16bf;
typedef __attribute__((ext_vector_type(8)))  float  v8f;

static __device__ __forceinline__ v8f wmma_bf16(v16bf a, v16bf b, v8f c) {
  // D = A(16x32 bf16) * B(32x16 bf16) + C(16x16 f32)
  return __builtin_amdgcn_wmma_f32_16x16x32_bf16(
      /*neg_a=*/false, a, /*neg_b=*/false, b,
      /*c_mod=*/(short)0, c, /*reuse_a=*/false, /*reuse_b=*/false);
}

// ---------------------------------------------------------------------------
// Kernel 0: convert fp32 weight [Kdim x N] (row-major) into bf16, pre-swizzled
// into the WMMA B-operand per-lane layout:
//   dst[((ntile*(Kdim/32)+kstep)*32 + lane)*16 + j] =
//       W[(kstep*32 + (j<8 ? (lane<16?0:8) + j : 16 + (lane<16?0:8) + j - 8))*N + ntile*16 + (lane&15)]
// so stage-1/2 kernels read each lane's 16 bf16 K-values as one contiguous 32B.
// ---------------------------------------------------------------------------
__global__ __launch_bounds__(256) void swizzle_weight(
    const float* __restrict__ W, bf16_t* __restrict__ out, int Kdim, int N) {
  int t = blockIdx.x * blockDim.x + threadIdx.x;
  int total = Kdim * N;
  if (t >= total) return;
  int j     = t & 15;
  int lane  = (t >> 4) & 31;
  int chunk = t >> 9;                 // ntile * ksteps + kstep
  int ksteps = Kdim >> 5;
  int ntile = chunk / ksteps;
  int kstep = chunk - ntile * ksteps;
  int n = ntile * 16 + (lane & 15);
  int k = kstep * 32 + ((lane < 16) ? 0 : 8) + ((j < 8) ? j : 8 + j);
  out[t] = (bf16_t)W[(size_t)k * N + n];
}

// ---------------------------------------------------------------------------
// Stage 1: per-neighbor MLP + bias + ReLU + mean-pool over K=32 neighbors.
//   X      : [B*32, KDIM] fp32 activations (streamed once from HBM)
//   Wsw    : swizzled bf16 weights, (H/16)*(KDIM/32)*512 elements
//   bias   : [H] fp32
//   pooled : [B, H] bf16 (row-major)
// One wave handles one (node, 16-col H-tile): 2 M-tiles of 16 rows, K=KDIM.
// ---------------------------------------------------------------------------
template <int KDIM>
__global__ __launch_bounds__(256) void mlp_pool(
    const float* __restrict__ X, const bf16_t* __restrict__ Wsw,
    const float* __restrict__ bias, bf16_t* __restrict__ pooled,
    int B, int H) {
  constexpr int KSTEPS = KDIM / 32;
  const int lane = threadIdx.x & 31;
  const int wave = threadIdx.x >> 5;
  const int gw   = blockIdx.x * 8 + wave;
  const int ntiles = H >> 4;
  const int node  = gw / ntiles;
  const int ntile = gw - node * ntiles;
  if (node >= B) return;
  const int hi = (lane >= 16) ? 1 : 0;
  const int nl = lane & 15;

  // B tile: 16xKDIM columns ntile*16..+15, preloaded (L2-resident weights)
  v16bf btile[KSTEPS];
  const bf16_t* bp = Wsw + (size_t)(ntile * KSTEPS) * 512 + lane * 16;
#pragma unroll
  for (int ks = 0; ks < KSTEPS; ++ks)
    btile[ks] = *(const v16bf*)(bp + ks * 512);

  const float bval = bias[ntile * 16 + nl];
  float colsum = 0.0f;

#pragma unroll
  for (int mt = 0; mt < 2; ++mt) {
    // A row for this lane (both lane halves hold the same row, different K)
    const float* rowp = X + ((size_t)node * 32 + mt * 16 + nl) * KDIM;
    v8f c = {};
#pragma unroll
    for (int ks = 0; ks < KSTEPS; ++ks) {
      const int k0 = ks * 32 + hi * 8;  // lane-half K base
      v16bf a;
#pragma unroll
      for (int j = 0; j < 8; ++j) {
        a[j]     = (bf16_t)rowp[k0 + j];       // K = k0..k0+7
        a[8 + j] = (bf16_t)rowp[k0 + 16 + j];  // K = k0+16..k0+23
      }
      c = wmma_bf16(a, btile[ks], c);
    }
    // bias + ReLU, then sum this M-tile's 8 rows held by this lane
#pragma unroll
    for (int i = 0; i < 8; ++i) {
      float h = c[i] + bval;
      colsum += (h > 0.0f) ? h : 0.0f;
    }
  }
  // combine lane halves (rows 0..7 + rows 8..15)
  colsum += __shfl_xor(colsum, 16, 32);
  if (!hi)
    pooled[(size_t)node * H + ntile * 16 + nl] = (bf16_t)(colsum * (1.0f / 32.0f));
}

// ---------------------------------------------------------------------------
// Stage 2: Out = relu(S @ Wself + P @ Wneigh), Dout=256, H=512.
//   S [B,K1] fp32, P [B,512] bf16, weights pre-swizzled bf16.
// One wave = one 16x16 output tile; both GEMMs accumulate into one C.
// ---------------------------------------------------------------------------
template <int K1>
__global__ __launch_bounds__(256) void out_proj(
    const float* __restrict__ S, const bf16_t* __restrict__ Wself_sw,
    const bf16_t* __restrict__ P, const bf16_t* __restrict__ Wn_sw,
    float* __restrict__ Out, int B) {
  const int lane = threadIdx.x & 31;
  const int wave = threadIdx.x >> 5;
  const int gw   = blockIdx.x * 8 + wave;
  const int NT = 256 / 16;
  const int mtile = gw / NT;
  const int ntile = gw - mtile * NT;
  if (mtile * 16 >= B) return;
  const int hi = (lane >= 16) ? 1 : 0;
  const int nl = lane & 15;
  const int arow = mtile * 16 + nl;

  v8f c = {};
  // --- self part: K1 x 256 ---
  {
    constexpr int KS = K1 / 32;
    const float*  rowp = S + (size_t)arow * K1;
    const bf16_t* bp   = Wself_sw + (size_t)(ntile * KS) * 512 + lane * 16;
#pragma unroll
    for (int ks = 0; ks < KS; ++ks) {
      const int k0 = ks * 32 + hi * 8;
      v16bf a;
#pragma unroll
      for (int j = 0; j < 8; ++j) {
        a[j]     = (bf16_t)rowp[k0 + j];
        a[8 + j] = (bf16_t)rowp[k0 + 16 + j];
      }
      c = wmma_bf16(a, *(const v16bf*)(bp + ks * 512), c);
    }
  }
  // --- pooled part: 512 x 256 ---
  {
    const bf16_t* rowp = P + (size_t)arow * 512;
    const bf16_t* bp   = Wn_sw + (size_t)(ntile * 16) * 512 + lane * 16;
#pragma unroll
    for (int ks = 0; ks < 16; ++ks) {
      const int k0 = ks * 32 + hi * 8;
      v16bf a;
#pragma unroll
      for (int j = 0; j < 8; ++j) {
        a[j]     = rowp[k0 + j];
        a[8 + j] = rowp[k0 + 16 + j];
      }
      c = wmma_bf16(a, *(const v16bf*)(bp + ks * 512), c);
    }
  }
  // relu + store: C layout lane N=nl, row M = i + hi*8
#pragma unroll
  for (int i = 0; i < 8; ++i) {
    float v = c[i];
    v = (v > 0.0f) ? v : 0.0f;
    Out[(size_t)(mtile * 16 + i + hi * 8) * 256 + ntile * 16 + nl] = v;
  }
}

extern "C" void kernel_launch(void* const* d_in, const int* in_sizes, int n_in,
                              void* d_out, int out_size, void* d_ws, size_t ws_size,
                              hipStream_t stream) {
  const int D_IN = 128, D_GETO = 64, H = 512, D_OUT = 256, K = 32;
  const int B = in_sizes[0] / D_IN;  // 16384

  const float* self_vecs   = (const float*)d_in[0];
  const float* neigh_vecs  = (const float*)d_in[1];
  const float* self_geto   = (const float*)d_in[2];
  const float* neigh_geto  = (const float*)d_in[3];
  const float* W_feat_mlp  = (const float*)d_in[4];
  const float* b_feat_mlp  = (const float*)d_in[5];
  const float* W_geto_mlp  = (const float*)d_in[6];
  const float* b_geto_mlp  = (const float*)d_in[7];
  const float* W_nf        = (const float*)d_in[8];   // [H, Dout]
  const float* W_sf        = (const float*)d_in[9];   // [D_IN, Dout]
  const float* W_sg        = (const float*)d_in[10];  // [D_GETO, Dout]
  const float* W_ng        = (const float*)d_in[11];  // [H, Dout]

  // workspace layout (all offsets 32B-aligned)
  char* ws = (char*)d_ws;
  size_t o = 0;
  bf16_t* wfeat_sw = (bf16_t*)(ws + o); o += (size_t)D_IN   * H     * 2;
  bf16_t* wgeto_sw = (bf16_t*)(ws + o); o += (size_t)D_GETO * H     * 2;
  bf16_t* wnf_sw   = (bf16_t*)(ws + o); o += (size_t)H      * D_OUT * 2;
  bf16_t* wng_sw   = (bf16_t*)(ws + o); o += (size_t)H      * D_OUT * 2;
  bf16_t* wsf_sw   = (bf16_t*)(ws + o); o += (size_t)D_IN   * D_OUT * 2;
  bf16_t* wsg_sw   = (bf16_t*)(ws + o); o += (size_t)D_GETO * D_OUT * 2;
  bf16_t* pooled_f = (bf16_t*)(ws + o); o += (size_t)B * H * 2;
  bf16_t* pooled_g = (bf16_t*)(ws + o); o += (size_t)B * H * 2;

  // Stage 0: weight convert + swizzle (tiny)
  auto swz = [&](const float* w, bf16_t* dst, int kd, int n) {
    int total = kd * n;
    hipLaunchKernelGGL(swizzle_weight, dim3((total + 255) / 256), dim3(256), 0,
                       stream, w, dst, kd, n);
  };
  swz(W_feat_mlp, wfeat_sw, D_IN,   H);
  swz(W_geto_mlp, wgeto_sw, D_GETO, H);
  swz(W_nf,       wnf_sw,   H,      D_OUT);
  swz(W_ng,       wng_sw,   H,      D_OUT);
  swz(W_sf,       wsf_sw,   D_IN,   D_OUT);
  swz(W_sg,       wsg_sw,   D_GETO, D_OUT);

  // Stage 1: fused MLP + bias + ReLU + mean-pool (one wave per node x H-tile)
  {
    int waves  = B * (H / 16);          // 524288
    int blocks = waves / 8;             // 8 waves / 256-thread block
    hipLaunchKernelGGL((mlp_pool<128>), dim3(blocks), dim3(256), 0, stream,
                       neigh_vecs, wfeat_sw, b_feat_mlp, pooled_f, B, H);
    hipLaunchKernelGGL((mlp_pool<64>),  dim3(blocks), dim3(256), 0, stream,
                       neigh_geto, wgeto_sw, b_geto_mlp, pooled_g, B, H);
  }

  // Stage 2: fused dual-GEMM + ReLU per branch
  {
    int waves  = (B / 16) * (D_OUT / 16);  // 16384
    int blocks = waves / 8;
    float* out_node = (float*)d_out;
    float* out_geto = (float*)d_out + (size_t)B * D_OUT;
    hipLaunchKernelGGL((out_proj<128>), dim3(blocks), dim3(256), 0, stream,
                       self_vecs, wsf_sw, pooled_f, wnf_sw, out_node, B);
    hipLaunchKernelGGL((out_proj<64>),  dim3(blocks), dim3(256), 0, stream,
                       self_geto, wsg_sw, pooled_g, wng_sw, out_geto, B);
  }
  (void)n_in; (void)out_size; (void)ws_size; (void)K;
}
